// SESConv_H_H_2731599200417
// MI455X (gfx1250) — matrix operand order repaired and checked by reading, not compiled, not run on hardware
//
#include <hip/hip_runtime.h>
#include <hip/hip_bf16.h>
#include <stdint.h>

// ---------------------------------------------------------------------------
// SESConv_H_H on MI455X (gfx1250): implicit-GEMM grouped conv via f16 WMMA.
//
//   out[b,o,g,h,w] = sum_{ij,c,x,y} Kern[g,o,c,ij,x,y] * xpad[b, s'(g,ij), h+x, w+y]
//   Kern[g,o,c,ij,x,y] = sum_f weight[o,c,ij,f] * basis[f,g,x,y]
//
// GEMM per (b,g,h-row): M=16 (o), N=128 (w, 8 WMMA tiles), K=1600 = 50 pairs
// of (slot, c16), each pair = one v_wmma_f32_16x16x32_f16.
// ---------------------------------------------------------------------------

typedef __attribute__((ext_vector_type(16))) _Float16 v16h;
typedef __attribute__((ext_vector_type(8)))  _Float16 v8h;
typedef __attribute__((ext_vector_type(8)))  float    v8f;

#define BB   8      // batch
#define CC   16     // in/out channels per group
#define SS   12     // NR*NS groups
#define HH   128
#define WW   128
#define HP   132    // H + 2*PAD padded
#define WP   132
#define NSLOT 100   // 4 (ij) * 25 (5x5)
#define NSP   50    // slot pairs (K=32 per WMMA)

// d_ws layout:
//   [0, 614400)              : Apack, uint32 [12][50][32 lanes][8 dwords]
//   [614400, 614400+57.99MB) : xpad f16 [b][13][132][132][16]  (plane 12 = zeros)
#define APACK_BYTES (12 * NSP * 32 * 8 * 4)

// ---------------------------------------------------------------------------
// Kernel 0: x (f32, [b][c][s][h][w]) -> xpad (f16, [b][s][hp][wp][c]),
// zero halo of 2 on h/w, plus an all-zero 13th s-plane for invalid wrap slots.
// ---------------------------------------------------------------------------
__global__ void build_xpad(const float* __restrict__ x, _Float16* __restrict__ xh) {
  const int NPIX = BB * 13 * HP * WP;
  int idx = blockIdx.x * blockDim.x + threadIdx.x;
  if (idx >= NPIX) return;
  int wp = idx % WP;  int t = idx / WP;
  int hp = t % HP;    t /= HP;
  int s  = t % 13;    int b = t / 13;

  _Float16 vals[16];
  bool inside = (s < SS) && (hp >= 2) && (hp < HH + 2) && (wp >= 2) && (wp < WW + 2);
  if (inside) {
    const float* src = x + ((long)(b * CC) * SS + s) * (HH * WW)
                         + (hp - 2) * WW + (wp - 2);
    #pragma unroll
    for (int c = 0; c < 16; ++c)
      vals[c] = (_Float16)src[(long)c * SS * HH * WW];
  } else {
    #pragma unroll
    for (int c = 0; c < 16; ++c) vals[c] = (_Float16)0.f;
  }
  _Float16* dst = xh + (long)idx * 16;
  *(v8h*)dst       = *(v8h*)vals;
  *(v8h*)(dst + 8) = *(v8h*)(vals + 8);
}

// ---------------------------------------------------------------------------
// Kernel 1: fuse weight x basis einsum, emit f16 A-fragments in a contiguous
// 16-bit A 16x32 per-lane k order (B loads use the same):
//   lane l: M = l%16, half = l/16
//   VGPR v : K = 16*half + 2v + t
//   K<16 -> slot 2sp (c=K), K>=16 -> slot 2sp+1 (c=K-16)
// slot = ij*25 + x*5 + y
// ---------------------------------------------------------------------------
__global__ void pack_A(const float* __restrict__ weight, const float* __restrict__ basis,
                       uint32_t* __restrict__ apack) {
  const int N = 12 * NSP * 32 * 8;
  int idx = blockIdx.x * blockDim.x + threadIdx.x;
  if (idx >= N) return;
  int v    = idx & 7;   int t = idx >> 3;
  int lane = t & 31;    t >>= 5;
  int sp   = t % NSP;   int g = t / NSP;
  int o    = lane & 15, half = lane >> 4;

  uint32_t packed = 0;
  #pragma unroll
  for (int tt = 0; tt < 2; ++tt) {
    int k    = 16 * half + 2 * v + tt;
    int slot = 2 * sp + (k >> 4);
    int c    = k & 15;
    int ij   = slot / 25, rem = slot % 25, xx = rem / 5, yy = rem % 5;
    float acc = 0.f;
    #pragma unroll
    for (int f = 0; f < 9; ++f)
      acc += weight[((o * 16 + c) * 4 + ij) * 9 + f]
           * basis[((f * 12 + g) * 5 + xx) * 5 + yy];
    _Float16 hv = (_Float16)acc;
    packed |= ((uint32_t)__builtin_bit_cast(uint16_t, hv)) << (16 * tt);
  }
  apack[idx] = packed;
}

// ---------------------------------------------------------------------------
// Kernel 2: one wave per (b,g,h). 8 accumulator tiles cover the 128-px row.
// B-fragment (16-bit 32x16): lanes 0-15 carry K=0..15 (slot A), lanes 16-31
// carry K=16..31 (slot B) -> each lane loads one contiguous 16-channel f16
// pixel vector (32B). EXEC stays all-ones (no divergence) as WMMA requires.
// ---------------------------------------------------------------------------
__global__ void __launch_bounds__(256)
ses_conv_wmma(const _Float16* __restrict__ xh,
              const uint32_t* __restrict__ apack,
              float* __restrict__ out) {
  int wave = (blockIdx.x * blockDim.x + threadIdx.x) >> 5;
  int lane = threadIdx.x & 31;
  int h = wave & 127;  int t = wave >> 7;
  int g = t % 12;      int b = t / 12;
  int n = lane & 15, half = lane >> 4;

  // source channel plane for each ij (12 = zero plane)
  int sch[4];
  #pragma unroll
  for (int ij = 0; ij < 4; ++ij) {
    int i = ij >> 1, j = ij & 1;
    int rr = ((g / 3) + i) & 3;
    int ss = (g % 3) + j;
    sch[ij] = (ss < 3) ? (rr * 3 + ss) : 12;
  }

  v8f acc[8];
  #pragma unroll
  for (int i = 0; i < 8; ++i)
    #pragma unroll
    for (int e = 0; e < 8; ++e) acc[i][e] = 0.f;

  const uint32_t* aptr = apack + ((long)g * NSP * 32 + lane) * 8;
  const long planeBase = (long)b * 13 * HP * WP * 16;

  #pragma unroll 1
  for (int sp = 0; sp < NSP; ++sp) {
    v16h afrag = *(const v16h*)aptr;
    __builtin_prefetch((const void*)(aptr + 32 * 8), 0, 1);  // next sp's A
    aptr += 32 * 8;

    // this lane's K-half selects slot 2sp (half=0) or 2sp+1 (half=1)
    int slot = 2 * sp + half;
    int ij = slot / 25, rem = slot % 25, xx = rem / 5, yy = rem % 5;
    const _Float16* bp = xh + planeBase
                       + ((long)sch[ij] * HP + (h + xx)) * (WP * 16)
                       + (n + yy) * 16;
    #pragma unroll
    for (int nt = 0; nt < 8; ++nt) {
      v16h bfrag = *(const v16h*)(bp + nt * 256);  // 16 px * 16 c per tile
      acc[nt] = __builtin_amdgcn_wmma_f32_16x16x32_f16(
          false, afrag, false, bfrag, (short)0, acc[nt], false, false);
    }
  }

  // D layout: lane -> N = lane%16; VGPR r -> M = r + 8*(lane/16)
  const long obase = ((long)(b * 16) * 12 + g) * (HH * WW) + h * WW;
  #pragma unroll
  for (int nt = 0; nt < 8; ++nt)
    #pragma unroll
    for (int r = 0; r < 8; ++r) {
      int o = r + 8 * half;
      out[obase + (long)o * 12 * HH * WW + nt * 16 + n] = acc[nt][r];
    }
}

// ---------------------------------------------------------------------------
extern "C" void kernel_launch(void* const* d_in, const int* in_sizes, int n_in,
                              void* d_out, int out_size, void* d_ws, size_t ws_size,
                              hipStream_t stream) {
  const float* x      = (const float*)d_in[0];   // (8,16,12,128,128) f32
  const float* weight = (const float*)d_in[1];   // (16,16,4,9) f32
  const float* basis  = (const float*)d_in[2];   // (9,12,5,5) f32
  float* out          = (float*)d_out;           // (8,16,12,128,128) f32

  uint32_t* apack = (uint32_t*)d_ws;
  _Float16* xh    = (_Float16*)((char*)d_ws + APACK_BYTES);  // ~58 MB needed total

  {
    int npix = BB * 13 * HP * WP;
    build_xpad<<<(npix + 255) / 256, 256, 0, stream>>>(x, xh);
  }
  {
    int n = 12 * NSP * 32 * 8;
    pack_A<<<(n + 255) / 256, 256, 0, stream>>>(weight, basis, apack);
  }
  {
    int nwaves = BB * 12 * HH;                    // 12288 waves
    ses_conv_wmma<<<nwaves * 32 / 256, 256, 0, stream>>>(xh, apack, out);
  }
}
